// Conv2d_NN_spatial_55362128445640
// MI455X (gfx1250) — matrix operand order repaired and checked by reading, not compiled, not run on hardware
//
#include <hip/hip_runtime.h>
#include <stdint.h>

// ---------------- problem constants (from reference) ----------------
#define B_   16
#define C_   128
#define H_   64
#define W_   64
#define N_   4096          // H*W tokens
#define M_   256           // SAMPLES^2
#define KNN  9
#define OUT_ 128
#define CK   (C_*KNN)      // 1152 = reduction dim of conv GEMM (ordered k*C + c)
#define IDX_STRIDE 16      // padded idx row

typedef __attribute__((ext_vector_type(16))) _Float16 v16h;
typedef __attribute__((ext_vector_type(8)))  float    v8f;
typedef __attribute__((ext_vector_type(4)))  uint32_t v4u;

union Frag16 { v16h h; v4u u[2]; };

// ---------------------------------------------------------------
// Kernel 1: gather static sample grid -> xs[b][m][c] (f16, sample-
// row-major so WMMA B fragments are contiguous), plus exact m2.
// ---------------------------------------------------------------
__global__ __launch_bounds__(128) void k_prep_sample(
    const float* __restrict__ x, _Float16* __restrict__ xs,
    float* __restrict__ m2)
{
    int blk = blockIdx.x;            // B_*M_ blocks
    int b = blk >> 8, m = blk & 255;
    int ix = m >> 4, iy = m & 15;
    // xi = round(linspace(0, 63, 16)) -- rintf matches round-half-even
    int px = (int)rintf((float)ix * (float)(H_ - 1) / 15.0f);
    int py = (int)rintf((float)iy * (float)(W_ - 1) / 15.0f);
    int pix = px * W_ + py;
    int c = threadIdx.x;             // 128 threads = C_
    float v = x[((size_t)(b * C_ + c)) * (size_t)N_ + pix];
    xs[((size_t)(b * M_ + m)) * C_ + c] = (_Float16)v;

    __shared__ float red[128];
    red[c] = v * v;
    __syncthreads();
    for (int s = 64; s > 0; s >>= 1) {
        if (c < s) red[c] += red[c + s];
        __syncthreads();
    }
    if (c == 0) m2[b * M_ + m] = red[0];
}

// ---------------------------------------------------------------
// Kernel 2: repack w(OUT,C,9) -> wBT[o][k*C + c] in f16 so the conv
// GEMM's B^T rows are contiguous and A-rows are 9 contiguous sample
// rows of 128 channels.
// ---------------------------------------------------------------
__global__ void k_prep_w(const float* __restrict__ w, _Float16* __restrict__ wBT)
{
    int e = blockIdx.x * blockDim.x + threadIdx.x;
    if (e >= OUT_ * CK) return;
    int o  = e / CK;
    int kk = e - o * CK;             // kk = k*C_ + c
    int k  = kk >> 7;
    int c  = kk & 127;
    wBT[e] = (_Float16)w[(o * C_ + c) * KNN + k];
}

// ---------------------------------------------------------------
// Kernel 3: distance GEMM (WMMA f16->f32) + top-9 selection.
// Block = 16 tokens, 128 threads (4 waves). Wave w owns sample
// tiles 4w..4w+3 (64 samples). dist2 = n2 + m2 - 2*dot in LDS,
// then 9-pass argmin per token.
// ---------------------------------------------------------------
__global__ __launch_bounds__(128) void k_dist_topk(
    const float* __restrict__ x, const _Float16* __restrict__ xs,
    const float* __restrict__ m2, int* __restrict__ idx)
{
    __shared__ __attribute__((aligned(16))) _Float16 la[16][C_]; // 4 KB token tile
    __shared__ float ln2[16];
    __shared__ float ld[16][M_];                                 // 16 KB dist tile

    int b    = blockIdx.x >> 8;
    int tok0 = (blockIdx.x & 255) << 4;
    int tid  = threadIdx.x;

    // stage A tile: la[token][c] (x is channel-major, stride N_)
    for (int e = tid; e < 16 * C_; e += 128) {
        int c = e >> 4, t = e & 15;
        la[t][c] = (_Float16)x[((size_t)(b * C_ + c)) * (size_t)N_ + tok0 + t];
    }
    __syncthreads();
    if (tid < 16) {                  // n2 from the same f16 data (consistent)
        float s = 0.f;
        for (int c = 0; c < C_; ++c) { float v = (float)la[tid][c]; s += v * v; }
        ln2[tid] = s;
    }
    __syncthreads();

    int wave = tid >> 5, lane = tid & 31;
    int row  = lane & 15;
    int hi   = (lane >> 4) & 1;

    v8f acc[4] = {};
    const _Float16* xsb = xs + (size_t)b * M_ * C_;

    for (int kb = 0; kb < C_ / 32; ++kb) {                 // 4 K-steps
        Frag16 a;
        int kbase = kb * 32 + hi * 16;                     // A 16-bit layout
        a.u[0] = *(const v4u*)&la[row][kbase];
        a.u[1] = *(const v4u*)&la[row][kbase + 8];
        int koff = kb * 32 + hi * 16;                      // B 16-bit layout
        for (int j = 0; j < 4; ++j) {
            int mcol = (wave * 4 + j) * 16 + row;
            Frag16 bb;
            const _Float16* p = xsb + (size_t)mcol * C_ + koff;
            bb.u[0] = *(const v4u*)p;
            bb.u[1] = *(const v4u*)(p + 8);
            acc[j] = __builtin_amdgcn_wmma_f32_16x16x32_f16(
                false, a.h, false, bb.h, (short)0, acc[j], false, false);
        }
    }

    // C/D layout: lane = N col (sample), VGPR r = M row (token), +8 for hi lanes
    for (int j = 0; j < 4; ++j) {
        int mcol  = (wave * 4 + j) * 16 + row;
        float m2v = m2[b * M_ + mcol];
        for (int r = 0; r < 8; ++r) {
            int t = r + hi * 8;
            ld[t][mcol] = ln2[t] + m2v - 2.0f * acc[j][r];
        }
    }
    __syncthreads();

    if (tid < 16) {                  // top-9 smallest per token
        int t = tid;
        int* op = idx + ((size_t)(b * N_ + tok0 + t)) * IDX_STRIDE;
        for (int kk = 0; kk < KNN; ++kk) {
            float best = ld[t][0]; int bi = 0;
            for (int m = 1; m < M_; ++m) {
                float v = ld[t][m];
                if (v < best) { best = v; bi = m; }
            }
            op[kk] = bi;
            ld[t][bi] = 3.0e38f;
        }
    }
}

// ---------------------------------------------------------------
// Kernel 4: gather + conv GEMM. Block = 16 tokens, 128 threads.
// A-panel (16 x 1152 f16, 36 KB LDS) = per token 9 contiguous
// sample rows. Wave w does OUT tiles {2w,2w+1}; 36 K-steps of
// wmma_f32_16x16x32_f16 each.
// ---------------------------------------------------------------
__global__ __launch_bounds__(128) void k_conv(
    const _Float16* __restrict__ xs, const _Float16* __restrict__ wBT,
    const int* __restrict__ idx, const float* __restrict__ bias,
    float* __restrict__ out)
{
    __shared__ __attribute__((aligned(16))) _Float16 la[16][CK];  // 36 KB
    __shared__ int lidx[16 * KNN];

    int b    = blockIdx.x >> 8;
    int tok0 = (blockIdx.x & 255) << 4;
    int tid  = threadIdx.x;

    for (int e = tid; e < 16 * KNN; e += 128) {
        int t = e / KNN, k = e - t * KNN;
        lidx[e] = idx[((size_t)(b * N_ + tok0 + t)) * IDX_STRIDE + k];
    }
    __syncthreads();

    // gather: A row(token) = concat of 9 sample rows (128 f16 each), as u32
    const uint32_t* xsb = (const uint32_t*)(xs + (size_t)b * M_ * C_);
    uint32_t* lau = (uint32_t*)&la[0][0];
    for (int e = tid; e < 16 * (CK / 2); e += 128) {       // 9216 u32, 72/thread
        int t   = e / (CK / 2);
        int rem = e - t * (CK / 2);                        // 0..575
        int k   = rem >> 6;                                // neighbor
        int cp  = rem & 63;                                // channel pair
        int m   = lidx[t * KNN + k];
        lau[t * (CK / 2) + rem] = xsb[m * (C_ / 2) + cp];
    }
    __syncthreads();

    int wave = tid >> 5, lane = tid & 31;
    int row  = lane & 15;
    int hi   = (lane >> 4) & 1;
    int ot0  = wave * 2;

    v8f acc[2] = {};
    for (int kb = 0; kb < CK / 32; ++kb) {                 // 36 K-steps
        Frag16 a;
        int kbase = kb * 32 + hi * 16;
        a.u[0] = *(const v4u*)&la[row][kbase];
        a.u[1] = *(const v4u*)&la[row][kbase + 8];
        int koff = kb * 32 + hi * 16;
        for (int j = 0; j < 2; ++j) {
            int o = (ot0 + j) * 16 + row;
            Frag16 bb;
            const _Float16* p = wBT + (size_t)o * CK + koff;
            bb.u[0] = *(const v4u*)p;
            bb.u[1] = *(const v4u*)(p + 8);
            acc[j] = __builtin_amdgcn_wmma_f32_16x16x32_f16(
                false, a.h, false, bb.h, (short)0, acc[j], false, false);
        }
    }

    // store: M = token (contiguous along n), N = out channel
    for (int j = 0; j < 2; ++j) {
        int o = (ot0 + j) * 16 + row;
        float bv = bias[o];
        float* op = out + ((size_t)(b * OUT_ + o)) * (size_t)N_ + tok0 + hi * 8;
        for (int r = 0; r < 8; ++r) op[r] = acc[j][r] + bv;
    }
}

// ---------------------------------------------------------------
extern "C" void kernel_launch(void* const* d_in, const int* in_sizes, int n_in,
                              void* d_out, int out_size, void* d_ws, size_t ws_size,
                              hipStream_t stream)
{
    (void)in_sizes; (void)n_in; (void)out_size; (void)ws_size;
    const float* x    = (const float*)d_in[0];   // (B,C,H,W) f32
    const float* w    = (const float*)d_in[1];   // (OUT,C,9) f32
    const float* bias = (const float*)d_in[2];   // (OUT,)    f32
    float* out = (float*)d_out;                  // (B,OUT,H,W) f32

    // workspace layout (offsets 256B-aligned)
    char* ws = (char*)d_ws;
    _Float16* xs  = (_Float16*)(ws);                       // 1,048,576 B
    float*    m2  = (float*)(ws + 1048576);                //    16,384 B
    _Float16* wBT = (_Float16*)(ws + 1064960);             //   294,912 B
    int*      idx = (int*)(ws + 1359872);                  // 4,194,304 B

    k_prep_sample<<<B_ * M_, 128, 0, stream>>>(x, xs, m2);
    k_prep_w<<<(OUT_ * CK + 255) / 256, 256, 0, stream>>>(w, wBT);
    k_dist_topk<<<B_ * (N_ / 16), 128, 0, stream>>>(x, xs, m2, idx);
    k_conv<<<B_ * (N_ / 16), 128, 0, stream>>>(xs, wBT, idx, bias, out);
}
